// RWRLayer_50148038148170
// MI455X (gfx1250) — matrix operand order repaired and checked by reading, not compiled, not run on hardware
//
#include <hip/hip_runtime.h>
#include <stdint.h>

typedef __attribute__((ext_vector_type(16))) _Float16 v16h;
typedef __attribute__((ext_vector_type(8)))  float    v8f;
typedef __attribute__((ext_vector_type(4)))  float    f32x4;
typedef __attribute__((ext_vector_type(4)))  uint32_t u32x4;
typedef __attribute__((ext_vector_type(8)))  uint32_t u32x8;

#define NN      8192
#define FIN     512
#define FOUT    64
#define ALPHA_  0.2f
#define JSPLIT  4
#define JCHUNK  (NN / JSPLIT)

union Frag { u32x8 u; v16h h; };

__device__ __forceinline__ uint32_t packh2(float lo, float hi) {
  union { _Float16 h[2]; uint32_t u; } un;
  un.h[0] = (_Float16)lo; un.h[1] = (_Float16)hi;
  return un.u;
}

__device__ __forceinline__ v8f wmma_f16(v16h a, v16h b, v8f c) {
  // v_wmma_f32_16x16x32_f16: (neg_a, A, neg_b, B, c_mod, C, reuse_a, reuse_b)
  return __builtin_amdgcn_wmma_f32_16x16x32_f16(false, a, false, b, (short)0, c,
                                                false, false);
}

// Load one WMMA B fragment (pair-packed f16 layout, two b128 loads).
__device__ __forceinline__ v16h load_bfrag(const uint32_t* __restrict__ bp,
                                           int row64, int hs) {
  const u32x4* p = (const u32x4*)bp + (row64 << 2) + hs;
  Frag f;
  u32x4 lo = p[0];
  u32x4 hi = p[2];
  f.u[0] = lo[0]; f.u[1] = lo[1]; f.u[2] = lo[2]; f.u[3] = lo[3];
  f.u[4] = hi[0]; f.u[5] = hi[1]; f.u[6] = hi[2]; f.u[7] = hi[3];
  return f.h;
}

// ---------- 1) W -> pair-packed f16 B-layout ----------
__global__ void k_repack_W(const float* __restrict__ W, uint32_t* __restrict__ Wp) {
  int idx = blockIdx.x * 256 + threadIdx.x;      // 16384 = 256 k-pairs * 64 n
  int n   = idx & 63;
  int kpg = idx >> 6;                            // 0..255
  int k   = kpg << 1;
  int kt  = k >> 5;
  int kp  = (k >> 1) & 15;
  Wp[((kt * 64 + n) << 4) + kp] = packh2(W[k * FOUT + n], W[(k + 1) * FOUT + n]);
}

// ---------- 2) h = x @ W (WMMA f16, fp32 accumulate) ----------
__global__ void k_gemm_h(const float* __restrict__ x, const uint32_t* __restrict__ Wp,
                         float* __restrict__ h) {
  const int wid  = (blockIdx.x * 256 + threadIdx.x) >> 5;   // 0..511 row tiles
  const int lane = threadIdx.x & 31;
  const int hs   = lane >> 4;
  const int mrow = lane & 15;
  const int i0   = wid << 4;
  const int r    = i0 + mrow;
  const int kb   = hs << 3;                                 // A-frag K base (0 or 8)

  v8f zero = {};
  v8f acc[4] = {zero, zero, zero, zero};

  for (int k0 = 0; k0 < FIN; k0 += 32) {
    const f32x4* xr = (const f32x4*)(x + (size_t)r * FIN + k0 + kb);
    f32x4 f0 = xr[0], f1 = xr[1], f2 = xr[4], f3 = xr[5];   // K = kb..kb+7, kb+16..kb+23
    float xv[16] = {f0.x, f0.y, f0.z, f0.w, f1.x, f1.y, f1.z, f1.w,
                    f2.x, f2.y, f2.z, f2.w, f3.x, f3.y, f3.z, f3.w};
    v16h a;
#pragma unroll
    for (int e = 0; e < 16; ++e) a[e] = (_Float16)xv[e];

    const int kt = k0 >> 5;
#pragma unroll
    for (int t = 0; t < 4; ++t) {
      v16h b = load_bfrag(Wp, kt * 64 + (t << 4) + mrow, hs);
      acc[t] = wmma_f16(a, b, acc[t]);
    }
  }
#pragma unroll
  for (int t = 0; t < 4; ++t)
#pragma unroll
    for (int v = 0; v < 8; ++v)
      h[(size_t)(i0 + v + (hs << 3)) * FOUT + (t << 4) + mrow] = acc[t][v];
}

// ---------- 3) a1[i] = h_i . a[:64], a2[i] = h_i . a[64:] ----------
__global__ void k_a12(const float* __restrict__ h, const float* __restrict__ av,
                      float* __restrict__ a1, float* __restrict__ a2) {
  int i = blockIdx.x * 256 + threadIdx.x;        // 8192
  float s1 = 0.f, s2 = 0.f;
#pragma unroll 8
  for (int n = 0; n < FOUT; ++n) {
    float hv = h[(size_t)i * FOUT + n];
    s1 += hv * av[n];
    s2 += hv * av[FOUT + n];
  }
  a1[i] = s1;
  a2[i] = s2;
}

// ---------- 4) gmax = max(a2)  (upper bound for softmax shift) ----------
__global__ void k_max(const float* __restrict__ a2, float* __restrict__ gmax) {
  __shared__ float s[256];
  float m = -3.4e38f;
  for (int i = threadIdx.x; i < NN; i += 256) m = fmaxf(m, a2[i]);
  s[threadIdx.x] = m;
  __syncthreads();
  for (int o = 128; o > 0; o >>= 1) {
    if (threadIdx.x < o) s[threadIdx.x] = fmaxf(s[threadIdx.x], s[threadIdx.x + o]);
    __syncthreads();
  }
  if (threadIdx.x == 0) gmax[0] = s[0];
}

// ---------- 5) h -> pair-packed f16 B-layout per 32-row block ----------
__global__ void k_repack_h(const float* __restrict__ h, uint32_t* __restrict__ hp) {
  int idx = blockIdx.x * 256 + threadIdx.x;      // 262144 = 4096 j-pairs * 64 n
  int n   = idx & 63;
  int kpg = idx >> 6;                            // 0..4095
  int j   = kpg << 1;
  int jt  = j >> 5;
  int kp  = (j >> 1) & 15;
  hp[((jt * 64 + n) << 4) + kp] = packh2(h[(size_t)j * FOUT + n],
                                         h[(size_t)(j + 1) * FOUT + n]);
}

// ---------- 6) streaming masked-softmax attention: partial acc per j-split ----------
__global__ void k_attn(const float* __restrict__ adj, const uint32_t* __restrict__ hp,
                       const float* __restrict__ a1v, const float* __restrict__ a2v,
                       const float* __restrict__ gmaxp,
                       float* __restrict__ pacc, float* __restrict__ pl) {
  const int wid   = (blockIdx.x * 256 + threadIdx.x) >> 5;  // 0..2047
  const int lane  = threadIdx.x & 31;
  const int hs    = lane >> 4;
  const int mrow  = lane & 15;
  const int split = wid & (JSPLIT - 1);
  const int tile  = wid >> 2;                               // 0..511
  const int i0    = tile << 4;
  const int r     = i0 + mrow;
  const int kb    = hs << 3;

  const float a1r = a1v[r];
  float mb = a1r + gmaxp[0];
  mb = mb > 0.f ? mb : ALPHA_ * mb;                         // row-wise logit upper bound

  v8f zero = {};
  v8f acc0 = zero, acc1 = zero, acc2 = zero, acc3 = zero, ls = zero;
  v16h ones;
#pragma unroll
  for (int e = 0; e < 16; ++e) ones[e] = (_Float16)1.0f;

  const float* adjr = adj + (size_t)r * NN;
  const int jbeg = split * JCHUNK;
  const int jend = jbeg + JCHUNK;

  for (int j0 = jbeg; j0 < jend; j0 += 32) {
    // adj is read exactly once: stream it non-temporally (don't pollute L2)
    const f32x4* ar = (const f32x4*)(adjr + j0 + kb);
    f32x4 m0 = __builtin_nontemporal_load(ar + 0);
    f32x4 m1 = __builtin_nontemporal_load(ar + 1);
    f32x4 m2 = __builtin_nontemporal_load(ar + 4);
    f32x4 m3 = __builtin_nontemporal_load(ar + 5);
    __builtin_prefetch(adjr + j0 + 256, 0, 0);

    const f32x4* sp = (const f32x4*)(a2v + j0 + kb);
    f32x4 s0 = sp[0], s1 = sp[1], s2 = sp[4], s3 = sp[5];

    float mv[16] = {m0.x, m0.y, m0.z, m0.w, m1.x, m1.y, m1.z, m1.w,
                    m2.x, m2.y, m2.z, m2.w, m3.x, m3.y, m3.z, m3.w};
    float sv[16] = {s0.x, s0.y, s0.z, s0.w, s1.x, s1.y, s1.z, s1.w,
                    s2.x, s2.y, s2.z, s2.w, s3.x, s3.y, s3.z, s3.w};

    v16h A;
#pragma unroll
    for (int e = 0; e < 16; ++e) {
      float t = a1r + sv[e];
      t = t > 0.f ? t : ALPHA_ * t;                         // leaky relu
      float p = (mv[e] > 0.f) ? __expf(t - mb) : 0.f;       // mask -> 0 weight
      A[e] = (_Float16)p;
    }

    const int jt = j0 >> 5;
    v16h b0 = load_bfrag(hp, jt * 64 +  0 + mrow, hs);
    v16h b1 = load_bfrag(hp, jt * 64 + 16 + mrow, hs);
    v16h b2 = load_bfrag(hp, jt * 64 + 32 + mrow, hs);
    v16h b3 = load_bfrag(hp, jt * 64 + 48 + mrow, hs);

    acc0 = wmma_f16(A, b0, acc0);
    acc1 = wmma_f16(A, b1, acc1);
    acc2 = wmma_f16(A, b2, acc2);
    acc3 = wmma_f16(A, b3, acc3);
    ls   = wmma_f16(A, ones, ls);                           // row sums (softmax denom)
  }

  float* pa = pacc + (size_t)split * NN * FOUT;
#pragma unroll
  for (int v = 0; v < 8; ++v) {
    int row = i0 + v + (hs << 3);
    pa[(size_t)row * FOUT +  0 + mrow] = acc0[v];
    pa[(size_t)row * FOUT + 16 + mrow] = acc1[v];
    pa[(size_t)row * FOUT + 32 + mrow] = acc2[v];
    pa[(size_t)row * FOUT + 48 + mrow] = acc3[v];
  }
  if (mrow == 0) {
    float* plp = pl + split * NN;
#pragma unroll
    for (int v = 0; v < 8; ++v) plp[i0 + v + (hs << 3)] = ls[v];
  }
}

// ---------- 7) combine splits, normalize, ELU ----------
__global__ void k_final(const float* __restrict__ pacc, const float* __restrict__ pl,
                        float* __restrict__ out) {
  int idx = blockIdx.x * 256 + threadIdx.x;      // 524288
  int i = idx >> 6;
  float p = pacc[idx] + pacc[idx + 1 * NN * FOUT] +
            pacc[idx + 2 * NN * FOUT] + pacc[idx + 3 * NN * FOUT];
  float l = pl[i] + pl[i + NN] + pl[i + 2 * NN] + pl[i + 3 * NN];
  float v = p / fmaxf(l, 1e-30f);
  out[idx] = v > 0.f ? v : __expf(v) - 1.0f;     // ELU
}

extern "C" void kernel_launch(void* const* d_in, const int* in_sizes, int n_in,
                              void* d_out, int out_size, void* d_ws, size_t ws_size,
                              hipStream_t stream) {
  (void)in_sizes; (void)n_in; (void)out_size; (void)ws_size;
  const float* x   = (const float*)d_in[0];   // [8192, 512]
  const float* adj = (const float*)d_in[1];   // [8192, 8192]
  const float* W   = (const float*)d_in[2];   // [512, 64]
  const float* av  = (const float*)d_in[3];   // [128]
  float* out = (float*)d_out;                 // [8192, 64]

  char* ws = (char*)d_ws;
  float*    h    = (float*)   (ws);                                   // 2 MB
  uint32_t* hp   = (uint32_t*)(ws + (2u << 20));                      // 1 MB
  uint32_t* Wp   = (uint32_t*)(ws + (3u << 20));                      // 64 KB
  float*    a1   = (float*)   (ws + (3u << 20) + (64u  << 10));       // 32 KB
  float*    a2   = (float*)   (ws + (3u << 20) + (96u  << 10));       // 32 KB
  float*    gmax = (float*)   (ws + (3u << 20) + (128u << 10));       // pad to 1 KB
  float*    pacc = (float*)   (ws + (3u << 20) + (129u << 10));       // 8 MB
  float*    pl   = (float*)   (ws + (3u << 20) + (129u << 10) + (8u << 20)); // 128 KB

  k_repack_W<<<64,   256, 0, stream>>>(W, Wp);
  k_gemm_h  <<<64,   256, 0, stream>>>(x, Wp, h);
  k_a12     <<<32,   256, 0, stream>>>(h, av, a1, a2);
  k_max     <<<1,    256, 0, stream>>>(a2, gmax);
  k_repack_h<<<1024, 256, 0, stream>>>(h, hp);
  k_attn    <<<256,  256, 0, stream>>>(adj, hp, a1, a2, gmax, pacc, pl);
  k_final   <<<2048, 256, 0, stream>>>(pacc, pl, out);
}
